// Retain_13185549598730
// MI455X (gfx1250) — matrix operand order repaired and checked by reading, not compiled, not run on hardware
//
#include <hip/hip_runtime.h>
#include <math.h>

// ---------------- problem constants ----------------
constexpr int Bn = 128;    // batch
constexpr int Sn = 128;    // sequence
constexpr int Cn = 20;     // codes per visit
constexpr int Dn = 256;    // hidden
constexpr int MCn = 128;   // conv mid channels
constexpr int TEn = 128;   // timestep embedding dim
constexpr int Tn = 1000;   // diffusion steps
constexpr int BSn = Bn * Sn;          // 16384
constexpr long BSDn = (long)BSn * Dn; // 4194304

// ---------------- WMMA types / helpers ----------------
typedef __attribute__((ext_vector_type(16))) __bf16 v16bf;
typedef __attribute__((ext_vector_type(8)))  float  v8f;
typedef __attribute__((ext_vector_type(8)))  unsigned short v8us;
typedef __attribute__((ext_vector_type(16))) unsigned short v16us;

union ABf { v16bf v; v8us h[2]; v16us u; };

__device__ inline v8f wmma_bf16(v16bf a, v16bf b, v8f c) {
  return __builtin_amdgcn_wmma_f32_16x16x32_bf16(false, a, false, b, (short)0, c, false, false);
}

// A-matrix 16x32 bf16 tile (row-major source, ld elements/row).
// Layout: lanes 0-15 -> M=lane, K=0..15 ; lanes 16-31 -> M=lane-16, K=16..31 (matches load_b).
__device__ inline v16bf load_a(const unsigned short* base, int ld) {
  int lane = threadIdx.x & 31;
  const unsigned short* p = base + (long)(lane & 15) * ld + ((lane & 16) ? 16 : 0);
  ABf t;
  t.u = *(const v16us*)(p);
  return t.v;
}

// B-matrix 32x16 bf16 tile from N-major packed weights (row n contiguous in K, ld = K stride).
// Layout (contiguous k, same as load_a): lanes 0-15 -> N=lane, K=0..15 ; lanes 16-31 -> N=lane-16, K=16..31.
__device__ inline v16bf load_b(const unsigned short* baseT, int ld) {
  int lane = threadIdx.x & 31;
  const unsigned short* p = baseT + (long)(lane & 15) * ld + ((lane & 16) ? 16 : 0);
  ABf t;
  t.u = *(const v16us*)(p);
  return t.v;
}

__device__ inline unsigned short f2bf(float f) {
  unsigned u = __float_as_uint(f);
  u += 0x7FFFu + ((u >> 16) & 1u);   // round-to-nearest-even
  return (unsigned short)(u >> 16);
}

// ---------------- generic kernels ----------------
__global__ void k_zero(unsigned* p, long n) {
  long i = (long)blockIdx.x * blockDim.x + threadIdx.x;
  if (i < n) p[i] = 0u;
}

// dst[n*Kd + k] = bf16(src[off + n*sN + k*sK])  (pack weights bf16, N-major)
__global__ void k_convert(const float* src, unsigned short* dst, int Kd, long total,
                          long sN, long sK, long off) {
  long i = (long)blockIdx.x * blockDim.x + threadIdx.x;
  if (i >= total) return;
  long n = i / Kd, k = i % Kd;
  dst[i] = f2bf(src[off + n * sN + k * sK]);
}

// C[M,N] = A(bf16 MxK, lda) @ Bt(bf16 NxK, ldb) + bias ; act==1 -> tanh
// One wave computes a 16x64 tile: 4 accumulators share each A load (4x A reuse).
__global__ void k_gemm(const unsigned short* A, const unsigned short* Bt, const float* bias,
                       float* Cout, int K, int lda, int ldb, int ldc, int act) {
  int m0 = blockIdx.x << 4, n0 = blockIdx.y << 6;
  v8f zero = {0.f,0.f,0.f,0.f,0.f,0.f,0.f,0.f};
  v8f accs[4] = {zero, zero, zero, zero};
  const unsigned short* Ab = A + (long)m0 * lda;
  const unsigned short* Bb = Bt + (long)n0 * ldb;
  int lane = threadIdx.x & 31;
  const unsigned short* apf = Ab + (long)(lane & 15) * lda + ((lane & 16) ? 8 : 0);
  for (int k = 0; k < K; k += 32) {
    if (k + 64 < K) __builtin_prefetch(apf + k + 64, 0, 1);   // global_prefetch_b8
    v16bf a = load_a(Ab + k, lda);
#pragma unroll
    for (int j = 0; j < 4; ++j)
      accs[j] = wmma_bf16(a, load_b(Bb + (long)(j * 16) * ldb + k, ldb), accs[j]);
  }
  int nn = lane & 15, mb = (lane & 16) ? 8 : 0;
#pragma unroll
  for (int j = 0; j < 4; ++j) {
    int n = n0 + j * 16 + nn;
    float bi = bias ? bias[n] : 0.f;
    float* crow = Cout + (long)(m0 + mb) * ldc + n;
#pragma unroll
    for (int r = 0; r < 8; ++r) {
      float v = accs[j][r] + bi;
      if (act == 1) v = tanhf(v);
      crow[(long)r * ldc] = v;
    }
  }
}

// ---------------- model kernels ----------------
__global__ void k_alphas(const float* betas, float* alphas) {
  if (threadIdx.x == 0 && blockIdx.x == 0) {
    float p = 1.f;
    for (int i = 0; i < Tn; ++i) { p *= (1.f - betas[i]); alphas[i] = p; }
  }
}

__global__ void k_embed(const int* seqs, const float* emb, float* x, unsigned short* xbf) {
  int bs = blockIdx.x, d = threadIdx.x;
  const int* row = seqs + (long)bs * Cn;
  float acc = 0.f;
  for (int c = 0; c < Cn; ++c) acc += emb[(long)row[c] * Dn + d];
  x[(long)bs * Dn + d] = acc;
  xbf[(long)bs * Dn + d] = f2bf(acc);
}

// one GRU time step: hg = h@Wh (+bh), gates fused, writes seq/seqbf/hnew/hbfnext
__global__ void k_gru_step(const float* xg, int t,
                           const unsigned short* hbf, const float* hprev,
                           const unsigned short* WhT, const float* bh,
                           float* seq, unsigned short* seqbf,
                           float* hnew, unsigned short* hbfn) {
  int b0 = blockIdx.x << 4, d0 = blockIdx.y << 4;
  v8f aR = {0.f,0.f,0.f,0.f,0.f,0.f,0.f,0.f};
  v8f aZ = aR, aN = aR;
  for (int k = 0; k < Dn; k += 32) {
    v16bf a = load_a(hbf + (long)b0 * Dn + k, Dn);
    aR = wmma_bf16(a, load_b(WhT + ((long)(0 * Dn) + d0) * Dn + k, Dn), aR);
    aZ = wmma_bf16(a, load_b(WhT + ((long)(1 * Dn) + d0) * Dn + k, Dn), aZ);
    aN = wmma_bf16(a, load_b(WhT + ((long)(2 * Dn) + d0) * Dn + k, Dn), aN);
  }
  int lane = threadIdx.x & 31;
  int n = lane & 15, mb = (lane & 16) ? 8 : 0;
  int d = d0 + n;
  float bhr = bh[d], bhz = bh[Dn + d], bhn = bh[2 * Dn + d];
  for (int r = 0; r < 8; ++r) {
    int b = b0 + mb + r;
    const float* xrow = xg + ((long)b * Sn + t) * (3 * Dn);
    float xr = xrow[d], xz = xrow[Dn + d], xn = xrow[2 * Dn + d];
    float rg = 1.f / (1.f + __expf(-(xr + aR[r] + bhr)));
    float zg = 1.f / (1.f + __expf(-(xz + aZ[r] + bhz)));
    float nn = tanhf(xn + rg * (aN[r] + bhn));
    float hp = hprev[(long)b * Dn + d];
    float hN = (1.f - zg) * nn + zg * hp;
    seq[((long)b * Sn + t) * Dn + d] = hN;
    seqbf[((long)b * Sn + t) * Dn + d] = f2bf(hN);
    hnew[(long)b * Dn + d] = hN;
    hbfn[(long)b * Dn + d] = f2bf(hN);
  }
}

// visit_w = softmax_s( visit . Wva + bva )
__global__ void k_time_attn(const float* seq, const float* Wva, const float* bva, float* w) {
  int b = blockIdx.x, s = threadIdx.x;
  __shared__ float red[Sn];
  const float* row = seq + ((long)b * Sn + s) * Dn;
  float acc = bva[0];
  for (int d = 0; d < Dn; ++d) acc += row[d] * Wva[d];
  red[s] = acc; __syncthreads();
  for (int off = 64; off > 0; off >>= 1) { if (s < off) red[s] = fmaxf(red[s], red[s + off]); __syncthreads(); }
  float m = red[0]; __syncthreads();
  float e = __expf(acc - m);
  red[s] = e; __syncthreads();
  for (int off = 64; off > 0; off >>= 1) { if (s < off) red[s] += red[s + off]; __syncthreads(); }
  w[(long)b * Sn + s] = e / red[0];
}

// c[b,d] = sum_s w[b,s] * vv[b,s,d] * xs[b,s,d]
__global__ void k_context(const float* w, const float* vv, const float* xs, float* out) {
  int i = blockIdx.x * 256 + threadIdx.x;   // B*D
  int b = i / Dn, d = i % Dn;
  float acc = 0.f;
  for (int s = 0; s < Sn; ++s) {
    long r = (long)b * Sn + s;
    acc += w[r] * vv[r * Dn + d] * xs[r * Dn + d];
  }
  out[i] = acc;
}

__global__ void k_outhead(const float* c, const float* Wout, const float* bout, float* out) {
  int i = threadIdx.x;        // 256 = B*2
  int b = i >> 1, j = i & 1;
  float acc = bout[j];
  for (int d = 0; d < Dn; ++d) acc += c[(long)b * Dn + d] * Wout[d * 2 + j];
  out[i] = acc;
}

// cat[b,s] = [ x[b,0,:] , wh[b,s-1,:] ]  (bf16)
__global__ void k_prepcat(const unsigned short* xbf, const float* whf, unsigned short* cat) {
  long i = (long)blockIdx.x * 256 + threadIdx.x;
  if (i >= BSDn) return;
  int d = (int)(i % Dn);
  long bs = i / Dn;
  int s = (int)(bs % Sn);
  int b = (int)(bs / Sn);
  unsigned short* crow = cat + bs * (2 * Dn);
  crow[d] = xbf[(long)b * Sn * Dn + d];
  crow[Dn + d] = (s >= 1) ? f2bf(whf[(bs - 1) * Dn + d]) : (unsigned short)0;
}

// 64->2 head + softmax over the 2 logits
__global__ void k_attn2(const float* z, const float* W2, const float* b2, float* attn) {
  int bs = blockIdx.x, j = threadIdx.x;   // 64 threads
  __shared__ float p0[64], p1[64];
  float zv = z[(long)bs * 64 + j];
  p0[j] = zv * W2[j * 2 + 0];
  p1[j] = zv * W2[j * 2 + 1];
  __syncthreads();
  for (int off = 32; off > 0; off >>= 1) {
    if (j < off) { p0[j] += p0[j + off]; p1[j] += p1[j + off]; }
    __syncthreads();
  }
  if (j == 0) {
    float l0 = p0[0] + b2[0], l1 = p1[0] + b2[1];
    float m = fmaxf(l0, l1);
    float e0 = __expf(l0 - m), e1 = __expf(l1 - m);
    float inv = 1.f / (e0 + e1);
    attn[(long)bs * 2 + 0] = e0 * inv;
    attn[(long)bs * 2 + 1] = e1 * inv;
  }
}

__global__ void k_align(const float* x, const float* whf, const float* attn, float* aligned) {
  long i = (long)blockIdx.x * 256 + threadIdx.x;
  if (i >= BSDn) return;
  int d = (int)(i % Dn);
  long bs = i / Dn;
  int s = (int)(bs % Sn);
  int b = (int)(bs / Sn);
  float ek = x[(long)b * Sn * Dn + d];
  float v;
  if (s == 0) v = ek;
  else v = ek * attn[bs * 2] + whf[(bs - 1) * Dn + d] * attn[bs * 2 + 1];
  aligned[i] = v;
}

__global__ void k_tembin(const int* tsteps, unsigned short* tembin) {
  int b = blockIdx.x, j = threadIdx.x;   // 128
  float tv = (float)tsteps[b];
  int h = j & 63;
  float f = __expf(-logf(10000.f) * (float)h / 64.f);
  float arg = tv * f;
  float v = (j < 64) ? __cosf(arg) : __sinf(arg);
  tembin[(long)b * TEn + j] = f2bf(v);
}

// xt = aligned*sqrt(a) + noise*sqrt(1-a) ; transposed into padded conv input [b, 1+d, s]
__global__ void k_xt(const float* aligned, const float* noise, const float* alphas,
                     const int* tsteps, unsigned short* Ap) {
  long i = (long)blockIdx.x * 256 + threadIdx.x;
  if (i >= BSDn) return;
  int d = (int)(i % Dn);
  long bs = i / Dn;
  int s = (int)(bs % Sn);
  int b = (int)(bs / Sn);
  float a = alphas[tsteps[b]];
  float v = aligned[i] * sqrtf(a) + noise[i] * sqrtf(1.f - a);
  Ap[((long)b * (Dn + 2) + 1 + d) * Sn + s] = f2bf(v);
}

// conv1d (kernel 3, SAME) as 3 shift-accumulated WMMA GEMMs over padded transposed input.
// One wave computes a 16x64 output tile (4 accumulators reuse each A load, over 3 taps).
// mode 1: h = silu(y + bias + temb) -> padded bf16 activation for conv2
// mode 2: pred[b, n, d] = y + bias  (f32, (B,S,D) layout)
__global__ void k_conv(const unsigned short* Ap, const unsigned short* Kt,
                       const float* bias, const float* temb,
                       unsigned short* outA, float* outP,
                       int Cin, int Nn, int mode) {
  int bd = blockIdx.x;            // B * (D/16)
  int b = bd >> 4;                // D/16 == 16
  int d0 = (bd & 15) << 4;
  int n0 = blockIdx.y << 6;
  v8f zero = {0.f,0.f,0.f,0.f,0.f,0.f,0.f,0.f};
  v8f accs[4] = {zero, zero, zero, zero};
  for (int tau = 0; tau < 3; ++tau) {
    const unsigned short* Ab = Ap + ((long)b * (Dn + 2) + d0 + tau) * Cin;
    const unsigned short* Bb = Kt + ((long)tau * Nn + n0) * Cin;
    for (int k = 0; k < Cin; k += 32) {
      v16bf a = load_a(Ab + k, Cin);
#pragma unroll
      for (int j = 0; j < 4; ++j)
        accs[j] = wmma_bf16(a, load_b(Bb + (long)(j * 16) * Cin + k, Cin), accs[j]);
    }
  }
  int lane = threadIdx.x & 31;
  int nn = lane & 15;
  int mb = (lane & 16) ? 8 : 0;
#pragma unroll
  for (int j = 0; j < 4; ++j) {
    int n = n0 + j * 16 + nn;
    float bi = bias[n];
    for (int r = 0; r < 8; ++r) {
      int d = d0 + mb + r;
      float v = accs[j][r] + bi;
      if (mode == 1) {
        v += temb[(long)b * Nn + n];
        v = v / (1.f + __expf(-v));   // silu
        outA[((long)b * (Dn + 2) + 1 + d) * Nn + n] = f2bf(v);
      } else {
        outP[((long)b * Nn + n) * Dn + d] = v;
      }
    }
  }
}

__global__ void k_gen(const float* aligned, const float* noise, const float* pred,
                      float* gen, unsigned short* genbf) {
  long i = (long)blockIdx.x * 256 + threadIdx.x;
  if (i >= BSDn) return;
  float g = aligned[i] + noise[i] - pred[i];
  gen[i] = g;
  genbf[i] = f2bf(g);
}

__global__ void k_combine(const float* p1, const float* p2, const float* n1, const float* n2,
                          float* op, float* on) {
  long i = (long)blockIdx.x * 256 + threadIdx.x;
  if (i >= BSDn) return;
  op[i] = p1[i] + p2[i];
  on[i] = n1[i] + n2[i];
}

// ---------------- host orchestration ----------------
extern "C" void kernel_launch(void* const* d_in, const int* in_sizes, int n_in,
                              void* d_out, int out_size, void* d_ws, size_t ws_size,
                              hipStream_t stream) {
  (void)in_sizes; (void)n_in; (void)out_size; (void)ws_size;
  const int*   input_seqs = (const int*)d_in[0];
  const int*   t1     = (const int*)d_in[5];
  const int*   t2     = (const int*)d_in[6];
  const float* noise1 = (const float*)d_in[7];
  const float* noise2 = (const float*)d_in[8];
  const float* emb    = (const float*)d_in[9];
  const float* vWi = (const float*)d_in[10]; const float* vWh = (const float*)d_in[11];
  const float* vbi = (const float*)d_in[12]; const float* vbh = (const float*)d_in[13];
  const float* rWi = (const float*)d_in[14]; const float* rWh = (const float*)d_in[15];
  const float* rbi = (const float*)d_in[16]; const float* rbh = (const float*)d_in[17];
  const float* Wva = (const float*)d_in[18]; const float* bva = (const float*)d_in[19];
  const float* Wba = (const float*)d_in[20]; const float* bba = (const float*)d_in[21];
  const float* Wout= (const float*)d_in[22]; const float* bout= (const float*)d_in[23];
  const float* Whk = (const float*)d_in[24]; const float* bhk = (const float*)d_in[25];
  const float* W1  = (const float*)d_in[26]; const float* b1  = (const float*)d_in[27];
  const float* W2  = (const float*)d_in[28]; const float* b2  = (const float*)d_in[29];
  const float* betas=(const float*)d_in[30];
  const float* Wt  = (const float*)d_in[31]; const float* bt  = (const float*)d_in[32];
  const float* K1  = (const float*)d_in[33]; const float* cb1 = (const float*)d_in[34];
  const float* K2  = (const float*)d_in[35]; const float* cb2 = (const float*)d_in[36];
  float* dout = (float*)d_out;

  // -------- bump allocator over workspace --------
  char* wsb = (char*)d_ws;
  size_t off = 0;
  auto alloc = [&](size_t bytes) -> void* {
    void* p = wsb + off;
    off = (off + bytes + 255) & ~(size_t)255;
    return p;
  };
  float*          x       = (float*)alloc(BSDn * 4);
  unsigned short* xbf     = (unsigned short*)alloc(BSDn * 2);
  float*          xg      = (float*)alloc((long)BSn * 3 * Dn * 4);
  float*          hstate  = (float*)alloc((long)Bn * Dn * 4);
  unsigned short* hbf0    = (unsigned short*)alloc((long)Bn * Dn * 2);
  unsigned short* hbf1    = (unsigned short*)alloc((long)Bn * Dn * 2);
  float*          visit   = (float*)alloc(BSDn * 4);
  unsigned short* visitbf = (unsigned short*)alloc(BSDn * 2);
  float*          var     = (float*)alloc(BSDn * 4);
  unsigned short* varbf   = (unsigned short*)alloc(BSDn * 2);
  float*          varw    = (float*)alloc(BSDn * 4);
  float*          vw      = (float*)alloc((long)BSn * 4);
  float*          cbuf    = (float*)alloc((long)Bn * Dn * 4);
  float*          whf     = (float*)alloc(BSDn * 4);
  unsigned short* cat     = (unsigned short*)alloc(BSDn * 2 * 2);   // (B,S,2D) bf16
  float*          zb      = (float*)alloc((long)BSn * 64 * 4);
  float*          attn    = (float*)alloc((long)BSn * 2 * 4);
  float*          aligned = (float*)alloc(BSDn * 4);
  unsigned short* Ap1     = (unsigned short*)alloc((long)Bn * (Dn + 2) * Sn * 2);
  unsigned short* Ap2     = (unsigned short*)alloc((long)Bn * (Dn + 2) * MCn * 2);
  float*          pred1   = (float*)alloc(BSDn * 4);
  float*          pred2   = (float*)alloc(BSDn * 4);
  float*          gen1    = (float*)alloc(BSDn * 4);
  unsigned short* gen1bf  = (unsigned short*)alloc(BSDn * 2);
  float*          gen2    = (float*)alloc(BSDn * 4);
  unsigned short* gen2bf  = (unsigned short*)alloc(BSDn * 2);
  unsigned short* tembin  = (unsigned short*)alloc((long)Bn * TEn * 2);
  float*          temb    = (float*)alloc((long)Bn * MCn * 4);
  float*          alphas  = (float*)alloc((long)Tn * 4);
  unsigned short* vWiT = (unsigned short*)alloc((long)3 * Dn * Dn * 2);
  unsigned short* vWhT = (unsigned short*)alloc((long)3 * Dn * Dn * 2);
  unsigned short* rWiT = (unsigned short*)alloc((long)3 * Dn * Dn * 2);
  unsigned short* rWhT = (unsigned short*)alloc((long)3 * Dn * Dn * 2);
  unsigned short* WhkT = (unsigned short*)alloc((long)Dn * Dn * 2);
  unsigned short* WbaT = (unsigned short*)alloc((long)Dn * Dn * 2);
  unsigned short* W1T  = (unsigned short*)alloc((long)64 * 2 * Dn * 2);
  unsigned short* WtT  = (unsigned short*)alloc((long)TEn * MCn * 2);
  unsigned short* Kt1  = (unsigned short*)alloc((long)3 * MCn * Sn * 2);
  unsigned short* Kt2  = (unsigned short*)alloc((long)3 * Sn * MCn * 2);

  auto packW = [&](const float* src, unsigned short* dst, int Kd, int Nn, long sN, long sK, long o) {
    long tot = (long)Kd * Nn;
    k_convert<<<dim3((unsigned)((tot + 255) / 256)), 256, 0, stream>>>(src, dst, Kd, tot, sN, sK, o);
  };

  // -------- weight packing (bf16, N-major) --------
  packW(vWi, vWiT, Dn, 3 * Dn, 1, 3 * Dn, 0);
  packW(vWh, vWhT, Dn, 3 * Dn, 1, 3 * Dn, 0);
  packW(rWi, rWiT, Dn, 3 * Dn, 1, 3 * Dn, 0);
  packW(rWh, rWhT, Dn, 3 * Dn, 1, 3 * Dn, 0);
  packW(Whk, WhkT, Dn, Dn, 1, Dn, 0);
  packW(Wba, WbaT, Dn, Dn, 1, Dn, 0);
  packW(W1,  W1T,  2 * Dn, 64, 1, 64, 0);
  packW(Wt,  WtT,  TEn, MCn, 1, MCn, 0);
  for (int tau = 0; tau < 3; ++tau) {
    packW(K1, Kt1 + (long)tau * MCn * Sn, Sn, MCn, (long)Sn * 3, 3, tau);   // K1[o,i,tau]
    packW(K2, Kt2 + (long)tau * Sn * MCn, MCn, Sn, (long)MCn * 3, 3, tau);  // K2[n,i,tau]
  }
  k_alphas<<<1, 1, 0, stream>>>(betas, alphas);

  // -------- embedding gather+sum --------
  k_embed<<<BSn, Dn, 0, stream>>>(input_seqs, emb, x, xbf);

  // -------- GRU runner --------
  auto run_gru = [&](const unsigned short* xin, const unsigned short* WiT, const float* bii,
                     const unsigned short* WhT, const float* bhh,
                     float* seq, unsigned short* seqbf) {
    k_gemm<<<dim3(BSn / 16, (3 * Dn) / 64), 32, 0, stream>>>(xin, WiT, bii, xg, Dn, Dn, Dn, 3 * Dn, 0);
    k_zero<<<(Bn * Dn + 255) / 256, 256, 0, stream>>>((unsigned*)hstate, (long)Bn * Dn);
    k_zero<<<(Bn * Dn / 2 + 255) / 256, 256, 0, stream>>>((unsigned*)hbf0, (long)Bn * Dn / 2);
    k_zero<<<(Bn * Dn / 2 + 255) / 256, 256, 0, stream>>>((unsigned*)hbf1, (long)Bn * Dn / 2);
    unsigned short* hb[2] = {hbf0, hbf1};
    int cur = 0;
    for (int t = 0; t < Sn; ++t) {
      k_gru_step<<<dim3(Bn / 16, Dn / 16), 32, 0, stream>>>(xg, t, hb[cur], hstate, WhT, bhh,
                                                            seq, seqbf, hstate, hb[cur ^ 1]);
      cur ^= 1;
    }
  };

  run_gru(xbf, vWiT, vbi, vWhT, vbh, visit, visitbf);
  run_gru(xbf, rWiT, rbi, rWhT, rbh, var, varbf);

  // -------- context + output head --------
  k_time_attn<<<Bn, Sn, 0, stream>>>(visit, Wva, bva, vw);
  k_gemm<<<dim3(BSn / 16, Dn / 64), 32, 0, stream>>>(varbf, WbaT, bba, varw, Dn, Dn, Dn, Dn, 1);
  k_context<<<(Bn * Dn) / 256, 256, 0, stream>>>(vw, varw, x, cbuf);
  k_outhead<<<1, 256, 0, stream>>>(cbuf, Wout, bout, dout);

  const unsigned ewGrid = (unsigned)((BSDn + 255) / 256);
  long apWords1 = (long)Bn * (Dn + 2) * Sn / 2;   // bf16 buffer as u32 words
  long apWords2 = (long)Bn * (Dn + 2) * MCn / 2;

  // -------- align + diffusion branch --------
  auto run_branch = [&](const unsigned short* rnnbf, const int* tst, const float* noise,
                        float* pred, float* gen, unsigned short* genbf) {
    k_gemm<<<dim3(BSn / 16, Dn / 64), 32, 0, stream>>>(rnnbf, WhkT, bhk, whf, Dn, Dn, Dn, Dn, 0);
    k_prepcat<<<ewGrid, 256, 0, stream>>>(xbf, whf, cat);
    k_gemm<<<dim3(BSn / 16, 64 / 64), 32, 0, stream>>>(cat, W1T, b1, zb, 2 * Dn, 2 * Dn, 2 * Dn, 64, 1);
    k_attn2<<<BSn, 64, 0, stream>>>(zb, W2, b2, attn);
    k_align<<<ewGrid, 256, 0, stream>>>(x, whf, attn, aligned);
    k_tembin<<<Bn, TEn, 0, stream>>>(tst, tembin);
    k_gemm<<<dim3(Bn / 16, MCn / 64), 32, 0, stream>>>(tembin, WtT, bt, temb, TEn, TEn, TEn, MCn, 0);
    k_zero<<<(unsigned)((apWords1 + 255) / 256), 256, 0, stream>>>((unsigned*)Ap1, apWords1);
    k_xt<<<ewGrid, 256, 0, stream>>>(aligned, noise, alphas, tst, Ap1);
    k_zero<<<(unsigned)((apWords2 + 255) / 256), 256, 0, stream>>>((unsigned*)Ap2, apWords2);
    k_conv<<<dim3(Bn * (Dn / 16), MCn / 64), 32, 0, stream>>>(Ap1, Kt1, cb1, temb, Ap2, nullptr, Sn, MCn, 1);
    k_conv<<<dim3(Bn * (Dn / 16), Sn / 64), 32, 0, stream>>>(Ap2, Kt2, cb2, nullptr, nullptr, pred, MCn, Sn, 2);
    k_gen<<<ewGrid, 256, 0, stream>>>(aligned, noise, pred, gen, genbf);
  };

  run_branch(visitbf, t1, noise1, pred1, gen1, gen1bf);
  run_branch(varbf,   t2, noise2, pred2, gen2, gen2bf);

  // -------- generated-sequence GRUs (reuse visit/var buffers) --------
  run_gru(gen1bf, vWiT, vbi, vWhT, vbh, visit, visitbf);
  run_gru(gen2bf, rWiT, rbi, rWhT, rbh, var, varbf);

  k_time_attn<<<Bn, Sn, 0, stream>>>(visit, Wva, bva, vw);
  k_gemm<<<dim3(BSn / 16, Dn / 64), 32, 0, stream>>>(varbf, WbaT, bba, varw, Dn, Dn, Dn, Dn, 1);
  k_context<<<(Bn * Dn) / 256, 256, 0, stream>>>(vw, varw, gen1, cbuf);
  k_outhead<<<1, 256, 0, stream>>>(cbuf, Wout, bout, dout + Bn * 2);

  // -------- final combined outputs --------
  k_combine<<<ewGrid, 256, 0, stream>>>(pred1, pred2, noise1, noise2,
                                        dout + 2 * Bn * 2, dout + 2 * Bn * 2 + BSDn);
}
